// TransDecSoftModel_64390149701918
// MI455X (gfx1250) — matrix operand order repaired and checked by reading, not compiled, not run on hardware
//
#include <hip/hip_runtime.h>

typedef __attribute__((ext_vector_type(16))) _Float16 v16h;
typedef __attribute__((ext_vector_type(8)))  float    v8f;

#define Hh   512
#define Bb   1024
#define Tt   20
#define Ll   2
#define Vv   101
#define VPAD 128        // v2e GEMM K padded to multiple of 32
#define EPAD 128        // e2v GEMM N padded to multiple of 64
#define DFFd 2048
#define CACHE_STRIDE ((size_t)Tt * Bb * Hh)   // per-layer K/V cache elements

// ---------------------------------------------------------------------------
// GEMM: C[M,N] = act(A[M,K]f16 * W[N,K]f16^T + bias + bias2), f32 accumulate
// via v_wmma_f32_16x16x32_f16. Each wave computes a 32x64 output block
// (2 M-tiles x 4 N-tiles = 8 accumulators): 8 WMMA per 32-K step on
// 48 loaded dwords; A fragments reused 4x, B fragments 2x in registers.
// Requires M % 32 == 0, N % 64 == 0, K % 32 == 0.
// ---------------------------------------------------------------------------
__global__ __launch_bounds__(128) void gemm_wmma_kernel(
    const _Float16* __restrict__ A, const _Float16* __restrict__ W,
    const float* __restrict__ bias, const float* __restrict__ bias2,
    float* __restrict__ C, _Float16* __restrict__ C16,
    int M, int N, int Kd, int relu)
{
  const int wave = threadIdx.x >> 5;
  const int lane = threadIdx.x & 31;
  const int tilesN = N >> 6;                  // 64-column tile groups
  const int tile = blockIdx.x * 4 + wave;
  if (tile >= (M >> 5) * tilesN) return;      // wave-uniform: EXEC all-1 for WMMA
  const int tm = tile / tilesN, tn = tile % tilesN;
  const int half = lane >> 4;                 // lane group (0: lanes 0-15, 1: 16-31)
  const int l15  = lane & 15;

  const unsigned int* Arow0 = (const unsigned int*)(A + (size_t)(tm * 32 + l15) * Kd);
  const unsigned int* Arow1 = (const unsigned int*)(A + (size_t)(tm * 32 + 16 + l15) * Kd);
  const unsigned int* Wrow[4];
#pragma unroll
  for (int j = 0; j < 4; ++j)
    Wrow[j] = (const unsigned int*)(W + (size_t)(tn * 64 + j * 16 + l15) * Kd);

  v8f acc[2][4];
#pragma unroll
  for (int mi = 0; mi < 2; ++mi)
#pragma unroll
    for (int j = 0; j < 4; ++j) acc[mi][j] = (v8f){};

  const int ksteps = Kd >> 5;
  for (int kk = 0; kk < ksteps; ++kk) {
    const int kd = kk << 4;                   // dword index of k0 within the row
    union { v16h h; unsigned int u[8]; } a0, a1, b;
    // A 16x32 f16 fragment: v0..3 -> K=2v+8g ; v4..7 -> K=16+2(v-4)+8g
#pragma unroll
    for (int v = 0; v < 4; ++v) {
      a0.u[v]     = Arow0[kd + (half << 2) + v];
      a0.u[v + 4] = Arow0[kd + 8 + (half << 2) + v];
      a1.u[v]     = Arow1[kd + (half << 2) + v];
      a1.u[v + 4] = Arow1[kd + 8 + (half << 2) + v];
    }
    if (kk + 1 < ksteps) {                    // prefetch next K-slab
      __builtin_prefetch(Arow0 + kd + 16, 0, 3);
      __builtin_prefetch(Arow1 + kd + 16, 0, 3);
    }
#pragma unroll
    for (int j = 0; j < 4; ++j) {
      // B 32x16 f16 fragment: v0..3 -> K=2v+8g ; v4..7 -> K=16+2(v-4)+8g
#pragma unroll
      for (int v = 0; v < 8; ++v) b.u[v] = Wrow[j][kd + (half << 2) + (v & 3) + ((v >> 2) << 3)];
      if (kk + 1 < ksteps) __builtin_prefetch(Wrow[j] + kd + 16, 0, 3);
      acc[0][j] = __builtin_amdgcn_wmma_f32_16x16x32_f16(
          false, a0.h, false, b.h, (short)0, acc[0][j], false, false);
      acc[1][j] = __builtin_amdgcn_wmma_f32_16x16x32_f16(
          false, a1.h, false, b.h, (short)0, acc[1][j], false, false);
    }
  }
  // D layout: VGPR v -> M = v + 8*half, N = lane&15
#pragma unroll
  for (int j = 0; j < 4; ++j) {
    const int ncol = tn * 64 + j * 16 + l15;
    float badd = (bias ? bias[ncol] : 0.f) + (bias2 ? bias2[ncol] : 0.f);
#pragma unroll
    for (int mi = 0; mi < 2; ++mi) {
#pragma unroll
      for (int v = 0; v < 8; ++v) {
        const int m = tm * 32 + mi * 16 + half * 8 + v;
        float val = acc[mi][j][v] + badd;
        if (relu) val = fmaxf(val, 0.f);
        if (C)   C[(size_t)m * N + ncol] = val;
        if (C16) C16[(size_t)m * N + ncol] = (_Float16)val;
      }
    }
  }
}

// ---------------------------------------------------------------------------
// Incremental self-attention: append newest K/V (pos S-1), softmax over S keys.
// One block (8 waves) per batch element.
// ---------------------------------------------------------------------------
__global__ __launch_bounds__(256) void attn_kernel(
    const float* __restrict__ qkv,            // [B, 3H] (bias included)
    _Float16* __restrict__ Kc, _Float16* __restrict__ Vc,  // [T, B, H] f16
    float* __restrict__ outf, _Float16* __restrict__ out16,
    int S)
{
  const int b = blockIdx.x, t = threadIdx.x;
  const int wave = t >> 5, lane = t & 31;
  __shared__ float sc[32];
  __shared__ float aw[32];
  const float* qrow = qkv + (size_t)b * (3 * Hh);
  const size_t nbase = (size_t)(S - 1) * Bb * Hh + (size_t)b * Hh;
  for (int h = t; h < Hh; h += 256) {
    Kc[nbase + h] = (_Float16)qrow[Hh + h];
    Vc[nbase + h] = (_Float16)qrow[2 * Hh + h];
  }
  __syncthreads();
  for (int j = wave; j < S; j += 8) {
    const _Float16* krow = Kc + (size_t)j * Bb * Hh + (size_t)b * Hh;
    float p = 0.f;
    for (int h = lane; h < Hh; h += 32) p += qrow[h] * (float)krow[h];
    for (int off = 16; off; off >>= 1) p += __shfl_xor(p, off, 32);
    if (lane == 0) sc[j] = p * 0.04419417382415922f;   // 1/sqrt(512)
  }
  __syncthreads();
  if (t == 0) {
    float mx = -1e30f;
    for (int j = 0; j < S; ++j) mx = fmaxf(mx, sc[j]);
    float sum = 0.f;
    for (int j = 0; j < S; ++j) { float e = __expf(sc[j] - mx); aw[j] = e; sum += e; }
    float inv = 1.f / sum;
    for (int j = 0; j < S; ++j) aw[j] *= inv;
  }
  __syncthreads();
  for (int h = t; h < Hh; h += 256) {
    float accv = 0.f;
    for (int j = 0; j < S; ++j)
      accv += aw[j] * (float)Vc[(size_t)j * Bb * Hh + (size_t)b * Hh + h];
    outf[(size_t)b * Hh + h] = accv;
    out16[(size_t)b * Hh + h] = (_Float16)accv;
  }
}

// ---------------------------------------------------------------------------
// y = LayerNorm(x + r) * g + beta  (H=512 per row), writes f32 + f16 copies
// ---------------------------------------------------------------------------
__global__ __launch_bounds__(256) void add_ln_kernel(
    const float* __restrict__ x, const float* __restrict__ r,
    const float* __restrict__ g, const float* __restrict__ beta,
    float* __restrict__ y, _Float16* __restrict__ y16)
{
  const int b = blockIdx.x, t = threadIdx.x;
  __shared__ float red[256];
  const size_t base = (size_t)b * Hh;
  float v0 = x[base + t] + r[base + t];
  float v1 = x[base + 256 + t] + r[base + 256 + t];
  red[t] = v0 + v1;
  __syncthreads();
  for (int s = 128; s > 0; s >>= 1) { if (t < s) red[t] += red[t + s]; __syncthreads(); }
  const float mean = red[0] * (1.f / Hh);
  __syncthreads();
  red[t] = (v0 - mean) * (v0 - mean) + (v1 - mean) * (v1 - mean);
  __syncthreads();
  for (int s = 128; s > 0; s >>= 1) { if (t < s) red[t] += red[t + s]; __syncthreads(); }
  const float inv = rsqrtf(red[0] * (1.f / Hh) + 1e-5f);
  float o0 = (v0 - mean) * inv * g[t] + beta[t];
  float o1 = (v1 - mean) * inv * g[256 + t] + beta[256 + t];
  y[base + t] = o0;       y[base + 256 + t] = o1;
  y16[base + t] = (_Float16)o0; y16[base + 256 + t] = (_Float16)o1;
}

// ---------------------------------------------------------------------------
// Emit logits row to d_out, softmax -> padded f16 probs (next-step v2e input)
// ---------------------------------------------------------------------------
__global__ __launch_bounds__(128) void logits_softmax_kernel(
    const float* __restrict__ logits,         // [B, EPAD]
    float* __restrict__ out,                  // [B, V1]
    _Float16* __restrict__ probs16)           // [B, VPAD]
{
  const int b = blockIdx.x, t = threadIdx.x;
  __shared__ float red[128];
  float v = (t < Vv) ? logits[(size_t)b * EPAD + t] : -1e30f;
  if (t < Vv) out[(size_t)b * Vv + t] = v;
  red[t] = v;
  __syncthreads();
  for (int s = 64; s > 0; s >>= 1) { if (t < s) red[t] = fmaxf(red[t], red[t + s]); __syncthreads(); }
  const float mx = red[0];
  __syncthreads();
  float e = (t < Vv) ? __expf(v - mx) : 0.f;
  red[t] = e;
  __syncthreads();
  for (int s = 64; s > 0; s >>= 1) { if (t < s) red[t] += red[t + s]; __syncthreads(); }
  probs16[(size_t)b * VPAD + t] = (_Float16)(e / red[0]);
}

// ---------------------------------------------------------------------------
// Small setup kernels
// ---------------------------------------------------------------------------
__global__ void cvt_pad_kernel(const float* __restrict__ src, _Float16* __restrict__ dst,
                               int srcRows, int srcCols, int dstRows, int dstCols)
{
  size_t i = (size_t)blockIdx.x * 256 + threadIdx.x;
  if (i >= (size_t)dstRows * dstCols) return;
  int r = (int)(i / dstCols), c = (int)(i % dstCols);
  dst[i] = (r < srcRows && c < srcCols) ? (_Float16)src[(size_t)r * srcCols + c]
                                        : (_Float16)0.f;
}

__global__ void embed_kernel(const int* __restrict__ meanings, const float* __restrict__ emb,
                             float* __restrict__ srcf, _Float16* __restrict__ src16)
{
  const int b = blockIdx.x, t = threadIdx.x;
  for (int h = t; h < Hh; h += 256) {
    float s = 0.f;
    for (int m = 0; m < 5; ++m) {
      int idx = meanings[b * 5 + m] + m * 10;
      s += emb[(size_t)idx * Hh + h];
    }
    srcf[(size_t)b * Hh + h] = s;
    src16[(size_t)b * Hh + h] = (_Float16)s;
  }
}

__global__ void pe_kernel(float* __restrict__ pe)
{
  const int tp = blockIdx.x, t = threadIdx.x;
  for (int h = t; h < Hh; h += 256) {
    int j = h >> 1;
    float d = __expf(-(__logf(10000.f) / (float)Hh) * (float)(2 * j));
    float ang = (float)tp * d;
    pe[(size_t)tp * Hh + h] = (h & 1) ? __cosf(ang) : __sinf(ang);
  }
}

__global__ void init_probs_kernel(_Float16* __restrict__ probs16)
{
  size_t i = (size_t)blockIdx.x * 256 + threadIdx.x;
  if (i >= (size_t)Bb * VPAD) return;
  int v = (int)(i & (VPAD - 1));
  probs16[i] = (_Float16)((v == 100) ? 1.f : 0.f);   // one-hot start token
}

__global__ void pad_bias_kernel(const float* __restrict__ b, float* __restrict__ bp)
{
  int t = threadIdx.x;
  if (t < EPAD) bp[t] = (t < Vv) ? b[t] : 0.f;
}

// ---------------------------------------------------------------------------
// Host side
// ---------------------------------------------------------------------------
namespace {
struct Bump {
  char* base; size_t off;
  template <typename T> T* get(size_t n) {
    size_t a = (off + 255) & ~size_t(255);
    off = a + n * sizeof(T);
    return (T*)(base + a);
  }
};

inline void gemm(hipStream_t s, const _Float16* A, const _Float16* W,
                 const float* bias, const float* bias2,
                 float* C, _Float16* C16, int M, int N, int Kd, int relu)
{
  int tiles = (M / 32) * (N / 64);
  gemm_wmma_kernel<<<dim3((tiles + 3) / 4), dim3(128), 0, s>>>(
      A, W, bias, bias2, C, C16, M, N, Kd, relu);
}

inline void cvt(hipStream_t s, const float* src, _Float16* dst,
                int sr, int sc, int dr, int dc)
{
  size_t n = (size_t)dr * dc;
  cvt_pad_kernel<<<dim3((unsigned)((n + 255) / 256)), dim3(256), 0, s>>>(src, dst, sr, sc, dr, dc);
}
} // namespace

extern "C" void kernel_launch(void* const* d_in, const int* in_sizes, int n_in,
                              void* d_out, int out_size, void* d_ws, size_t ws_size,
                              hipStream_t stream)
{
  (void)in_sizes; (void)n_in; (void)out_size; (void)ws_size;
  const int*   meanings  = (const int*)  d_in[0];
  const float* emb_table = (const float*)d_in[1];
  const float* v2e_w = (const float*)d_in[2];  const float* v2e_b = (const float*)d_in[3];
  const float* e2v_w = (const float*)d_in[4];  const float* e2v_b = (const float*)d_in[5];
  const float* sa_in_w  = (const float*)d_in[6];  const float* sa_in_b  = (const float*)d_in[7];
  const float* sa_out_w = (const float*)d_in[8];  const float* sa_out_b = (const float*)d_in[9];
  const float* ca_in_w  = (const float*)d_in[10]; const float* ca_in_b  = (const float*)d_in[11];
  const float* ca_out_w = (const float*)d_in[12]; const float* ca_out_b = (const float*)d_in[13];
  const float* ff1_w = (const float*)d_in[14]; const float* ff1_b = (const float*)d_in[15];
  const float* ff2_w = (const float*)d_in[16]; const float* ff2_b = (const float*)d_in[17];
  const float* ln_g  = (const float*)d_in[18]; const float* ln_b  = (const float*)d_in[19];
  float* outp = (float*)d_out;

  Bump ws{(char*)d_ws, 0};
  // f16 weights (L2-resident working set)
  _Float16* wh_v2e    = ws.get<_Float16>((size_t)Hh * VPAD);
  _Float16* wh_e2v    = ws.get<_Float16>((size_t)EPAD * Hh);
  _Float16* wh_sa_in  = ws.get<_Float16>((size_t)Ll * 3 * Hh * Hh);
  _Float16* wh_sa_out = ws.get<_Float16>((size_t)Ll * Hh * Hh);
  _Float16* wh_ca_v   = ws.get<_Float16>((size_t)Ll * Hh * Hh);
  _Float16* wh_ca_out = ws.get<_Float16>((size_t)Ll * Hh * Hh);
  _Float16* wh_ff1    = ws.get<_Float16>((size_t)Ll * DFFd * Hh);
  _Float16* wh_ff2    = ws.get<_Float16>((size_t)Ll * Hh * DFFd);
  // activations / caches
  _Float16* src16   = ws.get<_Float16>((size_t)Bb * Hh);
  _Float16* vtmp16  = ws.get<_Float16>((size_t)Bb * Hh);
  _Float16* x16     = ws.get<_Float16>((size_t)Bb * Hh);
  _Float16* attn16  = ws.get<_Float16>((size_t)Bb * Hh);
  _Float16* h16     = ws.get<_Float16>((size_t)Bb * DFFd);
  _Float16* probs16 = ws.get<_Float16>((size_t)Bb * VPAD);
  _Float16* Kc      = ws.get<_Float16>((size_t)Ll * CACHE_STRIDE);
  _Float16* Vc      = ws.get<_Float16>((size_t)Ll * CACHE_STRIDE);
  float* pe     = ws.get<float>((size_t)Tt * Hh);
  float* srcf   = ws.get<float>((size_t)Bb * Hh);
  float* cc     = ws.get<float>((size_t)Ll * Bb * Hh);
  float* xf     = ws.get<float>((size_t)Bb * Hh);
  float* qkv    = ws.get<float>((size_t)Bb * 3 * Hh);
  float* attnf  = ws.get<float>((size_t)Bb * Hh);
  float* obuf   = ws.get<float>((size_t)Bb * Hh);
  float* ffbuf  = ws.get<float>((size_t)Bb * Hh);
  float* logits = ws.get<float>((size_t)Bb * EPAD);
  float* e2vbp  = ws.get<float>(EPAD);

  // ---- one-time precompute (deterministic; re-run every call) ----
  cvt(stream, v2e_w, wh_v2e, Hh, Vv, Hh, VPAD);              // pad K 101->128
  cvt(stream, e2v_w, wh_e2v, Vv, Hh, EPAD, Hh);              // pad N 101->128
  cvt(stream, sa_in_w,  wh_sa_in,  Ll * 3 * Hh, Hh, Ll * 3 * Hh, Hh);
  cvt(stream, sa_out_w, wh_sa_out, Ll * Hh, Hh, Ll * Hh, Hh);
  cvt(stream, ca_out_w, wh_ca_out, Ll * Hh, Hh, Ll * Hh, Hh);
  cvt(stream, ff1_w, wh_ff1, Ll * DFFd, Hh, Ll * DFFd, Hh);
  cvt(stream, ff2_w, wh_ff2, Ll * Hh, DFFd, Ll * Hh, DFFd);
  for (int l = 0; l < Ll; ++l)                               // V-slice of ca_in_w
    cvt(stream, ca_in_w + ((size_t)l * 3 * Hh + 2 * Hh) * Hh,
        wh_ca_v + (size_t)l * Hh * Hh, Hh, Hh, Hh, Hh);
  pad_bias_kernel<<<dim3(1), dim3(128), 0, stream>>>(e2v_b, e2vbp);
  pe_kernel<<<dim3(Tt), dim3(256), 0, stream>>>(pe);
  embed_kernel<<<dim3(Bb), dim3(256), 0, stream>>>(meanings, emb_table, srcf, src16);
  init_probs_kernel<<<dim3((Bb * VPAD + 255) / 256), dim3(256), 0, stream>>>(probs16);

  // cross-attn over length-1 memory collapses to a per-batch constant cc[l]
  for (int l = 0; l < Ll; ++l) {
    gemm(stream, src16, wh_ca_v + (size_t)l * Hh * Hh,
         ca_in_b + (size_t)l * 3 * Hh + 2 * Hh, nullptr,
         nullptr, vtmp16, Bb, Hh, Hh, 0);
    gemm(stream, vtmp16, wh_ca_out + (size_t)l * Hh * Hh,
         ca_out_b + (size_t)l * Hh, nullptr,
         cc + (size_t)l * Bb * Hh, nullptr, Bb, Hh, Hh, 0);
  }

  // ---- autoregressive decode with exact KV-cache ----
  for (int i = 0; i < Tt; ++i) {
    const int S = i + 1;
    // x = probs @ v2e_w.T + v2e_b + pe[i]
    gemm(stream, probs16, wh_v2e, v2e_b, pe + (size_t)i * Hh, xf, x16, Bb, Hh, VPAD, 0);
    for (int l = 0; l < Ll; ++l) {
      gemm(stream, x16, wh_sa_in + (size_t)l * 3 * Hh * Hh,
           sa_in_b + (size_t)l * 3 * Hh, nullptr, qkv, nullptr, Bb, 3 * Hh, Hh, 0);
      attn_kernel<<<dim3(Bb), dim3(256), 0, stream>>>(
          qkv, Kc + (size_t)l * CACHE_STRIDE, Vc + (size_t)l * CACHE_STRIDE,
          attnf, attn16, S);
      gemm(stream, attn16, wh_sa_out + (size_t)l * Hh * Hh,
           sa_out_b + (size_t)l * Hh, nullptr, obuf, nullptr, Bb, Hh, Hh, 0);
      add_ln_kernel<<<dim3(Bb), dim3(256), 0, stream>>>(
          xf, obuf, ln_g + (size_t)(l * 3 + 0) * Hh, ln_b + (size_t)(l * 3 + 0) * Hh, xf, x16);
      add_ln_kernel<<<dim3(Bb), dim3(256), 0, stream>>>(
          xf, cc + (size_t)l * Bb * Hh,
          ln_g + (size_t)(l * 3 + 1) * Hh, ln_b + (size_t)(l * 3 + 1) * Hh, xf, x16);
      gemm(stream, x16, wh_ff1 + (size_t)l * DFFd * Hh,
           ff1_b + (size_t)l * DFFd, nullptr, nullptr, h16, Bb, DFFd, Hh, 1);
      gemm(stream, h16, wh_ff2 + (size_t)l * Hh * DFFd,
           ff2_b + (size_t)l * Hh, nullptr, ffbuf, nullptr, Bb, Hh, DFFd, 0);
      add_ln_kernel<<<dim3(Bb), dim3(256), 0, stream>>>(
          xf, ffbuf, ln_g + (size_t)(l * 3 + 2) * Hh, ln_b + (size_t)(l * 3 + 2) * Hh, xf, x16);
    }
    gemm(stream, x16, wh_e2v, e2vbp, nullptr, logits, nullptr, Bb, EPAD, Hh, 0);
    logits_softmax_kernel<<<dim3(Bb), dim3(128), 0, stream>>>(
        logits, outp + (size_t)i * Bb * Vv, probs16);
  }
}
